// DetrLoss_1108101562908
// MI455X (gfx1250) — matrix operand order repaired and checked by reading, not compiled, not run on hardware
//
#include <hip/hip_runtime.h>
#include <math.h>

#define BQ 16
#define QN 300
#define CN 1001
#define QP 304      // Q padded to multiple of 16
#define KP 1024     // C padded to multiple of 32
#define EPSF 1e-7f
#define NTILE 19    // ceil(300/16)

typedef __attribute__((ext_vector_type(16))) _Float16 v16h;
typedef __attribute__((ext_vector_type(8)))  _Float16 v8h;
typedef __attribute__((ext_vector_type(8)))  float    v8f;

// ---------- helpers ----------
__device__ __forceinline__ float4 cxcywh_to_xyxy(float4 b) {
    return make_float4(b.x - 0.5f * b.z, b.y - 0.5f * b.w,
                       b.x + 0.5f * b.z, b.y + 0.5f * b.w);
}

__device__ __forceinline__ float giou_loss(float4 a, float4 b) {
    float x1 = fmaxf(a.x, b.x), y1 = fmaxf(a.y, b.y);
    float x2 = fminf(a.z, b.z), y2 = fminf(a.w, b.w);
    float inter = fmaxf(x2 - x1, 0.f) * fmaxf(y2 - y1, 0.f);
    float a1 = (a.z - a.x) * (a.w - a.y);
    float a2 = (b.z - b.x) * (b.w - b.y);
    float uni = a1 + a2 - inter;
    float iou = inter / (uni + EPSF);
    float cx1 = fminf(a.x, b.x), cy1 = fminf(a.y, b.y);
    float cx2 = fmaxf(a.z, b.z), cy2 = fmaxf(a.w, b.w);
    float areac = (cx2 - cx1) * (cy2 - cy1);
    float giou = iou - (areac - uni) / (areac + EPSF);
    return 1.f - giou;
}

// ---------- kernel 1: gt argmax (labels) + indicator ----------
__global__ void aux_kernel(const float* __restrict__ gt_class,
                           int* __restrict__ gt_arg, float* __restrict__ ind) {
    int t = blockIdx.x * blockDim.x + threadIdx.x;
    if (t >= BQ * QN) return;
    const float* row = gt_class + (size_t)t * CN;
    float best = row[0]; int bi = 0;
    for (int c = 1; c < CN; ++c) {
        float v = row[c];
        if (v > best) { best = v; bi = c; }
    }
    gt_arg[t] = bi;
    ind[t] = 1.f - row[CN - 1];
}

// ---------- kernel 2: f32 -> zero-padded f16 staging (304 x 1024 per batch) ----------
__global__ void convert_kernel(const float* __restrict__ src, _Float16* __restrict__ dst) {
    size_t idx = (size_t)blockIdx.x * blockDim.x + threadIdx.x;
    size_t total = (size_t)BQ * QP * KP;
    if (idx >= total) return;
    int k = (int)(idx % KP);
    int r = (int)((idx / KP) % QP);
    int b = (int)(idx / ((size_t)KP * QP));
    float v = 0.f;
    if (r < QN && k < CN) v = src[((size_t)b * QN + r) * CN + k];
    dst[idx] = (_Float16)v;
}

// ---------- kernel 3: WMMA class-cost GEMM (pred_class @ gt_class^T, exact via one-hot)
//            fused with L1 + GIoU + indicator epilogue -> cost matrix ----------
__global__ void cost_wmma_kernel(const _Float16* __restrict__ A,
                                 const _Float16* __restrict__ Bm,
                                 const float* __restrict__ pred_boxes,
                                 const float* __restrict__ gt_boxes,
                                 const float* __restrict__ ind,
                                 float* __restrict__ cost) {
    int lane = threadIdx.x;                 // 0..31, one wave per (it, jt) tile
    int jt   = blockIdx.y * 4 + threadIdx.y;
    if (jt >= NTILE) return;                // uniform per wave
    int it = blockIdx.x;
    int b  = blockIdx.z;
    int i0 = it * 16, j0 = jt * 16;
    int kg = lane >> 4;                     // K-group (0/1)
    int ln = lane & 15;

    const _Float16* Arow = A  + ((size_t)b * QP + (i0 + ln)) * KP;
    const _Float16* Brow = Bm + ((size_t)b * QP + (j0 + ln)) * KP;

    v8f acc = {};
    for (int k0 = 0; k0 < KP; k0 += 32) {
        // A 16x32 f16 layout: lane row = i0+ln; halfs 0..7 = K[16*kg .. 16*kg+7],
        // halfs 8..15 = K[16*kg+8 .. 16*kg+15]
        v8h alo = *(const v8h*)(Arow + k0 + 16 * kg);
        v8h ahi = *(const v8h*)(Arow + k0 + 16 * kg + 8);
        v16h a = __builtin_shufflevector(alo, ahi,
                                         0,1,2,3,4,5,6,7,8,9,10,11,12,13,14,15);
        // B 32x16 f16 layout: lane col = j0+ln; 16 contiguous halfs K[16*kg ..]
        v16h bf = *(const v16h*)(Brow + k0 + 16 * kg);
        acc = __builtin_amdgcn_wmma_f32_16x16x32_f16(
            false, a, false, bf, (short)0, acc, false, false);
    }

    // Epilogue: C/D layout: acc[v] -> M = 8*kg + v, N = ln
    int j = j0 + ln;
    if (j < QN) {
        float4 gb = ((const float4*)gt_boxes)[(size_t)b * QN + j];
        float4 gx = cxcywh_to_xyxy(gb);
        float w = ind[b * QN + j];
        for (int v = 0; v < 8; ++v) {
            int i = i0 + 8 * kg + v;
            if (i < QN) {
                float4 pb = ((const float4*)pred_boxes)[(size_t)b * QN + i];
                float l1 = fabsf(pb.x - gb.x) + fabsf(pb.y - gb.y) +
                           fabsf(pb.z - gb.z) + fabsf(pb.w - gb.w);
                float gl = giou_loss(cxcywh_to_xyxy(pb), gx);
                cost[((size_t)b * QN + i) * QN + j] = w * (acc[v] + l1 + gl);
            }
        }
    }
}

// ---------- kernel 4: Jonker-Volgenant LSA, one workgroup per batch ----------
#define LSA_T 256
__global__ void lsa_kernel(const float* __restrict__ cost, int* __restrict__ gt_idx) {
    __shared__ float u_s[QN + 1], v_s[QN + 1], minv[QN + 1];
    __shared__ int   p_s[QN + 1], way[QN + 1], used[QN + 1];
    __shared__ float rv[LSA_T];
    __shared__ int   ri[LSA_T];
    __shared__ int   s_j0;

    const int b = blockIdx.x;
    const int tid = threadIdx.x;
    const int n = QN;
    const float* cb = cost + (size_t)b * QN * QN;

    for (int j = tid; j <= n; j += LSA_T) { u_s[j] = 0.f; v_s[j] = 0.f; p_s[j] = 0; way[j] = 0; }
    __syncthreads();

    for (int i = 1; i <= n; ++i) {
        if (tid == 0) { p_s[0] = i; s_j0 = 0; }
        for (int j = tid; j <= n; j += LSA_T) { minv[j] = INFINITY; used[j] = 0; }
        __syncthreads();

        while (true) {
            if (tid == 0) used[s_j0] = 1;
            __syncthreads();
            int j0 = s_j0;
            int i0 = p_s[j0];
            float u0 = u_s[i0];
            for (int j = tid + 1; j <= n; j += LSA_T) {
                if (!used[j]) {
                    float cur = cb[(size_t)(i0 - 1) * QN + (j - 1)] - u0 - v_s[j];
                    if (cur < minv[j]) { minv[j] = cur; way[j] = j0; }
                }
            }
            __syncthreads();
            // arg-min over unused columns
            float best = INFINITY; int bidx = 1;
            for (int j = tid + 1; j <= n; j += LSA_T) {
                float m = used[j] ? INFINITY : minv[j];
                if (m < best) { best = m; bidx = j; }
            }
            rv[tid] = best; ri[tid] = bidx;
            __syncthreads();
            for (int s = LSA_T / 2; s > 0; s >>= 1) {
                if (tid < s && rv[tid + s] < rv[tid]) { rv[tid] = rv[tid + s]; ri[tid] = ri[tid + s]; }
                __syncthreads();
            }
            float delta = rv[0];
            int j1 = ri[0];
            // dual updates (p_s of used columns are distinct rows -> no races)
            for (int j = tid; j <= n; j += LSA_T) {
                if (used[j]) { u_s[p_s[j]] += delta; v_s[j] -= delta; }
                else         { minv[j] -= delta; }
            }
            __syncthreads();
            if (tid == 0) s_j0 = j1;
            __syncthreads();
            if (p_s[j1] == 0) break;
        }
        // augmenting path (sequential)
        if (tid == 0) {
            int jj = s_j0;
            while (jj != 0) { int jp = way[jj]; p_s[jj] = p_s[jp]; jj = jp; }
        }
        __syncthreads();
    }
    for (int j = tid + 1; j <= n; j += LSA_T)
        gt_idx[b * QN + (p_s[j] - 1)] = j - 1;
}

// ---------- kernel 5a: zero the output scalar ----------
__global__ void zero_kernel(float* out) { out[0] = 0.f; }

// ---------- kernel 5b: final loss (CE via block log-softmax + L1 + GIoU) ----------
__global__ void loss_kernel(const float* __restrict__ pred_class,
                            const float* __restrict__ pred_boxes,
                            const float* __restrict__ gt_boxes,
                            const int* __restrict__ gt_idx,
                            const int* __restrict__ gt_arg,
                            float* __restrict__ out) {
    __shared__ float sdata[256];
    int bi = blockIdx.x;              // row index over B*Q
    int b = bi / QN;
    int tid = threadIdx.x;
    const float* x = pred_class + (size_t)bi * CN;

    float m = -INFINITY;
    for (int c = tid; c < CN; c += 256) m = fmaxf(m, x[c]);
    sdata[tid] = m; __syncthreads();
    for (int s = 128; s > 0; s >>= 1) {
        if (tid < s) sdata[tid] = fmaxf(sdata[tid], sdata[tid + s]);
        __syncthreads();
    }
    float mx = sdata[0]; __syncthreads();

    float sum = 0.f;
    for (int c = tid; c < CN; c += 256) sum += expf(x[c] - mx);
    sdata[tid] = sum; __syncthreads();
    for (int s = 128; s > 0; s >>= 1) {
        if (tid < s) sdata[tid] += sdata[tid + s];
        __syncthreads();
    }

    if (tid == 0) {
        float lse = mx + logf(sdata[0]);
        int g = gt_idx[bi];
        int label = gt_arg[b * QN + g];
        float ce = -(x[label] - lse) * (1.f / (float)QN);
        float4 pb = ((const float4*)pred_boxes)[bi];
        float4 gb = ((const float4*)gt_boxes)[(size_t)b * QN + g];
        float l1 = fabsf(pb.x - gb.x) + fabsf(pb.y - gb.y) +
                   fabsf(pb.z - gb.z) + fabsf(pb.w - gb.w);
        float gl = giou_loss(cxcywh_to_xyxy(pb), cxcywh_to_xyxy(gb));
        atomicAdd(out, ce + l1 + gl);
    }
}

// ---------- launcher ----------
extern "C" void kernel_launch(void* const* d_in, const int* in_sizes, int n_in,
                              void* d_out, int out_size, void* d_ws, size_t ws_size,
                              hipStream_t stream) {
    (void)in_sizes; (void)n_in; (void)out_size; (void)ws_size;
    const float* pred_class = (const float*)d_in[0];
    const float* pred_boxes = (const float*)d_in[1];
    const float* gt_class   = (const float*)d_in[2];
    const float* gt_boxes   = (const float*)d_in[3];
    float* out = (float*)d_out;

    char* ws = (char*)d_ws;
    const size_t nAB = (size_t)BQ * QP * KP;                // f16 elems per staging buffer
    _Float16* Af16 = (_Float16*)ws;
    _Float16* Bf16 = (_Float16*)(ws + nAB * 2);
    float* cost    = (float*)(ws + nAB * 4);
    char* p        = ws + nAB * 4 + (size_t)BQ * QN * QN * 4;
    int*   gt_arg  = (int*)p;               p += (size_t)BQ * QN * 4;
    float* ind     = (float*)p;             p += (size_t)BQ * QN * 4;
    int*   gt_idx  = (int*)p;

    // 1) labels + indicator
    aux_kernel<<<(BQ * QN + 255) / 256, 256, 0, stream>>>(gt_class, gt_arg, ind);

    // 2) f16 staging (zero padded)
    int convBlocks = (int)((nAB + 255) / 256);
    convert_kernel<<<convBlocks, 256, 0, stream>>>(pred_class, Af16);
    convert_kernel<<<convBlocks, 256, 0, stream>>>(gt_class,   Bf16);

    // 3) WMMA cost matrix (4 waves per block, one 16x16 tile per wave)
    dim3 gg(NTILE, (NTILE + 3) / 4, BQ);
    dim3 gb(32, 4, 1);
    cost_wmma_kernel<<<gg, gb, 0, stream>>>(Af16, Bf16, pred_boxes, gt_boxes, ind, cost);

    // 4) Hungarian assignment, one workgroup per batch
    lsa_kernel<<<BQ, LSA_T, 0, stream>>>(cost, gt_idx);

    // 5) final scalar loss
    zero_kernel<<<1, 1, 0, stream>>>(out);
    loss_kernel<<<BQ * QN, 256, 0, stream>>>(pred_class, pred_boxes, gt_boxes,
                                             gt_idx, gt_arg, out);
}
